// Attention_88012469829745
// MI455X (gfx1250) — matrix operand order repaired and checked by reading, not compiled, not run on hardware
//
#include <hip/hip_runtime.h>
#include <hip/hip_bf16.h>
#include <math.h>

typedef __attribute__((ext_vector_type(16))) __bf16 v16bf;
typedef __attribute__((ext_vector_type(8)))  float  v8f;

#define B_   128
#define H_   16
#define W_   48
#define D_   512
#define U_   512
#define M_   (B_ * H_ * W_)      /* 98304 rows of the big GEMM */
#define RPB_ (H_ * W_)           /* 768 rows per batch element */

// ---- native gfx1250 tanh (1 trans op instead of ~20 VALU ops) -----------
static __device__ __forceinline__ float fast_tanh(float x) {
#if __has_builtin(__builtin_amdgcn_tanhf)
  return __builtin_amdgcn_tanhf(x);
#else
  float r;
  // v_nop fills the required independent-op slot after a TRANS op
  asm volatile("v_tanh_f32 %0, %1\n\tv_nop" : "=v"(r) : "v"(x));
  return r;
#endif
}

// ---- CDNA5 async global->LDS copy helpers (ASYNCcnt-tracked) ------------
static __device__ __forceinline__ void async_b128_to_lds(unsigned lds_addr,
                                                         unsigned long long gaddr) {
  asm volatile("global_load_async_to_lds_b128 %0, %1, off"
               :: "v"(lds_addr), "v"(gaddr) : "memory");
}
static __device__ __forceinline__ void wait_async0() {
  asm volatile("s_wait_asynccnt 0" ::: "memory");
}

// -------------------------------------------------------------------------
// Kernel 0: dbias[b][u] = dec_hidden[b,:] @ W2_w[:,u] + W2_b[u] + W1_b[u]
// -------------------------------------------------------------------------
__global__ void __launch_bounds__(256) dec_bias_kernel(
    const float* __restrict__ dec, const float* __restrict__ W2,
    const float* __restrict__ W2b, const float* __restrict__ W1b,
    float* __restrict__ dbias) {
  __shared__ float ds[D_];
  const int b = blockIdx.x;
  for (int i = threadIdx.x; i < D_; i += blockDim.x) ds[i] = dec[b * D_ + i];
  __syncthreads();
  for (int u = threadIdx.x; u < U_; u += blockDim.x) {
    float acc = W2b[u] + W1b[u];
#pragma unroll 8
    for (int d = 0; d < D_; ++d) acc += ds[d] * W2[d * U_ + u];   // coalesced over u
    dbias[b * U_ + u] = acc;
  }
}

// -------------------------------------------------------------------------
// Kernel 1: prepack W1_w (f32, K-major [d][u]) into bf16 B-matrix fragments.
// Dense 16-bit B (32x16): lane L: N = L&15, hi = L>>4 ; elem e -> K = 32*chunk + 16*hi + e
// One thread produces one 32-byte v16bf fragment; fid = (nt*16+chunk)*32 + lane
// -------------------------------------------------------------------------
__global__ void __launch_bounds__(256) pack_w1_kernel(
    const float* __restrict__ W1, v16bf* __restrict__ Bfrag) {
  const int fid   = blockIdx.x * blockDim.x + threadIdx.x;  // 0 .. 16383
  const int ntile = fid >> 9;
  const int rem   = fid & 511;
  const int chunk = rem >> 5;
  const int lane  = rem & 31;
  const int n     = ntile * 16 + (lane & 15);
  const int hi    = lane >> 4;
  const int kbase = chunk * 32 + hi * 16;
  v16bf f;
#pragma unroll
  for (int e = 0; e < 16; ++e) f[e] = (__bf16)W1[(kbase + e) * U_ + n];
  Bfrag[fid] = f;
}

// -------------------------------------------------------------------------
// Kernel 2: fused GEMM (M=98304, K=512, N=512, bf16 WMMA, f32 acc)
//           + bias + tanh + dot(V) epilogue -> scores[M]
// Block = 256 thr = 8 waves; each wave owns 16 rows; block = 128 rows = 1/6 of one b.
// B tiles (16 KB) staged once per block via double-buffered async copies to
// LDS (8x less L2 traffic than per-wave global loads).
// Inner loop: 2 independent accumulators (halved WMMA RAW chain) and B
// fragments preloaded one pair ahead (relaxes s_wait_dscnt 0 -> <=2).
// A frag (16-bit A 16x32): lane L: M=L&15, hi=L>>4; e0..15->K=16hi+e
// C/D: lane L: N=L&15; vgpr i holds M = i + 8*(L>>4)
// -------------------------------------------------------------------------
__global__ void __launch_bounds__(256) attn_score_kernel(
    const float* __restrict__ enc, const v16bf* __restrict__ Bfrag,
    const float* __restrict__ dbias, const float* __restrict__ Vw,
    const float* __restrict__ Vb, float* __restrict__ scores) {
  __shared__ float db_s[U_];
  __shared__ float vw_s[U_];
  __shared__ __align__(128) char Bs[2][16384];   // double-buffered B tile

  const int b = blockIdx.x / 6;                 // 768 rows per b = 6 blocks of 128
  for (int i = threadIdx.x; i < U_; i += blockDim.x) {
    db_s[i] = dbias[b * U_ + i];
    vw_s[i] = Vw[i];
  }

  const int wave = threadIdx.x >> 5;
  const int lane = threadIdx.x & 31;
  const int m    = lane & 15;
  const int hi   = lane >> 4;

  const long row  = (long)blockIdx.x * 128 + wave * 16 + m;
  const float* rowp = enc + row * (long)D_;

  // LDS byte offsets of the two staging buffers (addrspace(3) offset lives
  // in the low 32 bits of the generic pointer)
  const unsigned bs0 = (unsigned)(unsigned long long)(void*)&Bs[0][0];
  const unsigned bs1 = (unsigned)(unsigned long long)(void*)&Bs[1][0];

  // ---- kick off async staging of B tile 0 while we build A fragments ----
  {
    const unsigned long long src =
        (unsigned long long)((const char*)Bfrag) + threadIdx.x * 64ull;
    const unsigned dst = bs0 + threadIdx.x * 64u;
#pragma unroll
    for (int j = 0; j < 4; ++j) async_b128_to_lds(dst + j * 16, src + j * 16);
  }

  // ---- load this wave's 16 A fragments (K = 512 in 16 chunks of 32) ----
  v16bf af[16];
#pragma unroll
  for (int c = 0; c < 16; ++c) {
    const float4* p = (const float4*)(rowp + c * 32 + hi * 16);
    float4 a0 = p[0];
    float4 a1 = p[1];
    const float4* q = (const float4*)(rowp + c * 32 + hi * 16 + 8);
    float4 a2 = q[0];
    float4 a3 = q[1];
    v16bf f;
    f[0]  = (__bf16)a0.x; f[1]  = (__bf16)a0.y; f[2]  = (__bf16)a0.z; f[3]  = (__bf16)a0.w;
    f[4]  = (__bf16)a1.x; f[5]  = (__bf16)a1.y; f[6]  = (__bf16)a1.z; f[7]  = (__bf16)a1.w;
    f[8]  = (__bf16)a2.x; f[9]  = (__bf16)a2.y; f[10] = (__bf16)a2.z; f[11] = (__bf16)a2.w;
    f[12] = (__bf16)a3.x; f[13] = (__bf16)a3.y; f[14] = (__bf16)a3.z; f[15] = (__bf16)a3.w;
    af[c] = f;
  }

  float sc[8];
#pragma unroll
  for (int i = 0; i < 8; ++i) sc[i] = 0.f;

  // ---- N loop: 32 tiles of 16 columns of U; K loop: 16 WMMAs ----
  for (int nt = 0; nt < 32; ++nt) {
    wait_async0();        // this wave's staging issues are complete
    __syncthreads();      // all waves' staging for tile nt is complete

    if (nt + 1 < 32) {    // prefetch next tile into the other buffer
      const unsigned long long src = (unsigned long long)((const char*)Bfrag) +
                                     (unsigned long long)(nt + 1) * 16384ull +
                                     threadIdx.x * 64ull;
      const unsigned dst = ((nt + 1) & 1 ? bs1 : bs0) + threadIdx.x * 64u;
#pragma unroll
      for (int j = 0; j < 4; ++j) async_b128_to_lds(dst + j * 16, src + j * 16);
    }

    const v16bf* bp = (const v16bf*)&Bs[nt & 1][0];
    v8f acc0 = {}, acc1 = {};
    v16bf b0 = bp[lane];                 // preload chunk pair 0
    v16bf b1 = bp[32 + lane];
#pragma unroll
    for (int c = 0; c < 16; c += 2) {
      v16bf n0 = b0, n1 = b1;
      if (c + 2 < 16) {                  // preload next pair before issuing WMMAs
        n0 = bp[(c + 2) * 32 + lane];
        n1 = bp[(c + 3) * 32 + lane];
      }
      acc0 = __builtin_amdgcn_wmma_f32_16x16x32_bf16(
          false, af[c],     false, b0, (short)0, acc0, false, false);
      acc1 = __builtin_amdgcn_wmma_f32_16x16x32_bf16(
          false, af[c + 1], false, b1, (short)0, acc1, false, false);
      b0 = n0; b1 = n1;
    }
    // epilogue: tanh(acc + dbias[u]) * V_w[u], u = column handled by this lane
    const int   u  = nt * 16 + m;
    const float db = db_s[u];
    const float vw = vw_s[u];
#pragma unroll
    for (int i = 0; i < 8; ++i) sc[i] += fast_tanh(acc0[i] + acc1[i] + db) * vw;
  }

  // ---- reduce over the 16 lanes of each half (the N dimension) ----
#pragma unroll
  for (int i = 0; i < 8; ++i) {
    float v = sc[i];
    v += __shfl_xor(v, 1, 32);
    v += __shfl_xor(v, 2, 32);
    v += __shfl_xor(v, 4, 32);
    v += __shfl_xor(v, 8, 32);
    sc[i] = v;
  }
  const float vb = Vb[0];
  if (m == 0) {                                  // lanes 0 (M=0..7) and 16 (M=8..15)
    const long base = (long)blockIdx.x * 128 + wave * 16 + hi * 8;
#pragma unroll
    for (int i = 0; i < 8; ++i) scores[base + i] = sc[i] + vb;
  }
}

// -------------------------------------------------------------------------
// Kernel 3: softmax over H (axis=1) per (b,w), write attn weights, then
//           context[b,d] = sum_{h,w} attn * enc  (memory-bound pass)
// -------------------------------------------------------------------------
__global__ void __launch_bounds__(256) softmax_ctx_kernel(
    const float* __restrict__ enc, const float* __restrict__ scores,
    float* __restrict__ ctx, float* __restrict__ attn) {
  __shared__ float s_s[RPB_];
  __shared__ float p_s[RPB_];
  const int b = blockIdx.x;
  for (int i = threadIdx.x; i < RPB_; i += blockDim.x)
    s_s[i] = scores[b * RPB_ + i];
  __syncthreads();

  if (threadIdx.x < W_) {
    const int w = threadIdx.x;
    float mx = -INFINITY;
#pragma unroll
    for (int h = 0; h < H_; ++h) mx = fmaxf(mx, s_s[h * W_ + w]);
    float e[H_];
    float sum = 0.f;
#pragma unroll
    for (int h = 0; h < H_; ++h) { e[h] = __expf(s_s[h * W_ + w] - mx); sum += e[h]; }
    const float inv = 1.0f / sum;
#pragma unroll
    for (int h = 0; h < H_; ++h) {
      const float p = e[h] * inv;
      p_s[h * W_ + w] = p;
      attn[b * RPB_ + h * W_ + w] = p;
    }
  }
  __syncthreads();

  const int d0 = threadIdx.x;                    // 256 threads cover D=512 as 2 each
  float a0 = 0.f, a1 = 0.f;
  const float* eb = enc + (long)b * RPB_ * D_;
  for (int hw = 0; hw < RPB_; ++hw) {
    const float p = p_s[hw];
    const float* ep = eb + (long)hw * D_;
    a0 += p * ep[d0];
    a1 += p * ep[d0 + 256];
  }
  ctx[b * D_ + d0]       = a0;
  ctx[b * D_ + d0 + 256] = a1;
}

// -------------------------------------------------------------------------
extern "C" void kernel_launch(void* const* d_in, const int* in_sizes, int n_in,
                              void* d_out, int out_size, void* d_ws, size_t ws_size,
                              hipStream_t stream) {
  const float* dec = (const float*)d_in[0];
  const float* enc = (const float*)d_in[1];
  const float* W1w = (const float*)d_in[2];
  const float* W1b = (const float*)d_in[3];
  const float* W2w = (const float*)d_in[4];
  const float* W2b = (const float*)d_in[5];
  const float* Vw  = (const float*)d_in[6];
  const float* Vb  = (const float*)d_in[7];

  char*  ws     = (char*)d_ws;
  float* dbias  = (float*)(ws);                       // 128*512*4   = 256 KB
  v16bf* Bfrag  = (v16bf*)(ws + 262144);              // 16384*32B   = 512 KB
  float* scores = (float*)(ws + 262144 + 524288);     // 98304*4     = 384 KB

  float* ctx  = (float*)d_out;                        // 128*512 f32
  float* attn = (float*)d_out + B_ * D_;              // 128*16*48 f32

  dec_bias_kernel   <<<B_,       256, 0, stream>>>(dec, W2w, W2b, W1b, dbias);
  pack_w1_kernel    <<<64,       256, 0, stream>>>(W1w, Bfrag);
  attn_score_kernel <<<M_ / 128, 256, 0, stream>>>(enc, Bfrag, dbias, Vw, Vb, scores);
  softmax_ctx_kernel<<<B_,       256, 0, stream>>>(enc, scores, ctx, attn);
}
